// SpatialLocalAttention_35450660061972
// MI455X (gfx1250) — matrix operand order repaired and checked by reading, not compiled, not run on hardware
//
#include <hip/hip_runtime.h>
#include <hip/hip_bf16.h>

// SpatialLocalAttention for MI455X (gfx1250, wave32, WMMA f16 16x16x32).
//
// Pipeline (all f16 WMMA with f32 accumulate):
//   k0: pack conv weights f32 [O][I][3][3] -> f16 [O][tap][ch] (k = tap*C+ch)
//   k1: per-window QKV conv as GEMM  C[192x64] = Wqkv[192x288] x im2col[288x64]
//   k2: flash attention per (t,head): S=Q*Kt (K=512), online softmax, O=P*V
//   k3: full-image 3x3 conv 64->64 as GEMM (K=576) + residual + bias
//
// Fragment layouts (CDNA5 ISA 7.12.2, wave32):
//   A 16x32 f16 : lane L -> row M=L%16; halves e: K = (e&8)*2 + (L/16)*8 + (e&7)
//                 -> two 16B chunks at kbase+hi*8 and kbase+16+hi*8
//   B 32x16 f16 : lane L -> col N=L%16; halves e: K = (e&8)*2 + (L/16)*8 + (e&7)
//                 -> two 16B chunks at kbase+hi*8 and kbase+16+hi*8
//   C/D 16x16 f32: VGPR r -> M = r + 8*(L/16), N = L%16

typedef __attribute__((ext_vector_type(16))) _Float16 v16h;
typedef __attribute__((ext_vector_type(8)))  _Float16 v8h;
typedef __attribute__((ext_vector_type(8)))  float    v8f;
typedef __attribute__((ext_vector_type(4)))  int      v4i;

#define T_DIM  8
#define C_IN   32
#define C_OUT  64
#define N_HEAD 8
#define WSZ    8
#define IMG    256
#define NWIN   1024   // 32x32 windows per t
#define DDIM   512    // per-head token dim = 8 ch * 64 px
#define WPB    8      // windows per block in k1

#if __has_builtin(__builtin_amdgcn_global_load_async_to_lds_b128)
#define HAVE_ASYNC_LDS 1
typedef __attribute__((address_space(1))) v4i* gptr_v4i;
typedef __attribute__((address_space(3))) v4i* lptr_v4i;
#else
#define HAVE_ASYNC_LDS 0
#endif

__device__ __forceinline__ v8f vzero8() {
  v8f z = {0.f,0.f,0.f,0.f,0.f,0.f,0.f,0.f};
  return z;
}

// Stage one 16B chunk from global into LDS. On CDNA5 use the ASYNCcnt-tracked
// GLOBAL_LOAD_ASYNC_TO_LDS_B128 path (no VGPR round trip); else plain copy.
__device__ __forceinline__ void stage16(const _Float16* g, _Float16* l) {
#if HAVE_ASYNC_LDS
  __builtin_amdgcn_global_load_async_to_lds_b128(
      (gptr_v4i)(g), (lptr_v4i)(l), 0, 0);
#else
  *(uint4*)l = *(const uint4*)g;
#endif
}

__device__ __forceinline__ void stage_wait() {
#if HAVE_ASYNC_LDS
#if __has_builtin(__builtin_amdgcn_s_wait_asynccnt)
  __builtin_amdgcn_s_wait_asynccnt(0);
#else
  asm volatile("s_wait_asynccnt 0x0" ::: "memory");
#endif
#endif
}

__device__ __forceinline__ v16h load_afrag(const _Float16* row, int kbase, int hi) {
  union { v16h v; v8h h[2]; } u;
  u.h[0] = *reinterpret_cast<const v8h*>(row + kbase + hi * 8);
  u.h[1] = *reinterpret_cast<const v8h*>(row + kbase + 16 + hi * 8);
  return u.v;
}

__device__ __forceinline__ v16h load_bfrag(const _Float16* row, int kbase, int hi) {
  union { v16h v; v8h h[2]; } u;
  u.h[0] = *reinterpret_cast<const v8h*>(row + kbase + hi * 8);
  u.h[1] = *reinterpret_cast<const v8h*>(row + kbase + 16 + hi * 8);
  return u.v;
}

__device__ __forceinline__ v8f wmma16(v16h a, v16h b, v8f c) {
  return __builtin_amdgcn_wmma_f32_16x16x32_f16(false, a, false, b, (short)0, c,
                                                false, false);
}

// ---------------------------------------------------------------- k0: weights
__global__ void k0_pack_weights(const float* Wq, const float* Wk, const float* Wv,
                                const float* Wo, _Float16* wqkv, _Float16* woh) {
  int i = blockIdx.x * blockDim.x + threadIdx.x;
  const int NQKV = 192 * 288;
  if (i < NQKV) {
    int m = i / 288, k = i % 288;
    int tap = k / 32, ch = k % 32;
    int p = m / 64, co = m % 64;
    const float* W = (p == 0) ? Wq : (p == 1) ? Wk : Wv;
    wqkv[i] = (_Float16)W[(co * C_IN + ch) * 9 + tap];
  } else if (i < NQKV + 64 * 576) {
    int j = i - NQKV;
    int co = j / 576, k = j % 576;
    int tap = k / 64, ch = k % 64;
    woh[j] = (_Float16)Wo[(co * C_OUT + ch) * 9 + tap];
  }
}

// ------------------------------------------------------------- k1: QKV convs
// 384 threads = 12 waves; wave w owns output rows [w*16, w*16+16) of M=192.
// LDS: As[192][288] f16 (110.6KB) + halo[10][10][32] f16 (6.4KB)
__global__ void k1_qkv(const float* __restrict__ x,
                       const float* __restrict__ bq, const float* __restrict__ bk,
                       const float* __restrict__ bv,
                       const _Float16* __restrict__ wqkv,
                       _Float16* __restrict__ qh, _Float16* __restrict__ kh,
                       _Float16* __restrict__ vth) {
  extern __shared__ char smem_raw[];
  _Float16* As = (_Float16*)smem_raw;      // [192][288]
  _Float16* halo = As + 192 * 288;         // [10][10][32]
  const int tid = threadIdx.x;
  const int lane = tid & 31, ln = lane & 15, hi = lane >> 4;
  const int wave = tid >> 5;  // 0..11

  // stage packed weights once per block (async on CDNA5)
  for (int c = tid; c < 192 * 288 / 8; c += 384)
    stage16(wqkv + c * 8, As + c * 8);
  stage_wait();

  for (int w = 0; w < WPB; ++w) {
    int win = blockIdx.x * WPB + w;
    int t = win >> 10;
    int rem = win & 1023;          // token id within (t)
    int wy = rem >> 5, wx = rem & 31;
    __syncthreads();
    for (int i = tid; i < 10 * 10 * 32; i += 384) halo[i] = (_Float16)0.f;
    __syncthreads();
    for (int i = tid; i < C_IN * 64; i += 384) {
      int ch = i >> 6, pix = i & 63;
      int py = pix >> 3, px = pix & 7;
      float v = x[(((size_t)t * C_IN + ch) * IMG + wy * 8 + py) * IMG + wx * 8 + px];
      halo[((py + 1) * 10 + (px + 1)) * 32 + ch] = (_Float16)v;
      if (w + 1 < WPB) {  // prefetch next window's elements -> global_prefetch_b8
        int nwin = win + 1;
        int nt_ = nwin >> 10, nrem = nwin & 1023;
        int nwy = nrem >> 5, nwx = nrem & 31;
        __builtin_prefetch(
            &x[(((size_t)nt_ * C_IN + ch) * IMG + nwy * 8 + py) * IMG + nwx * 8 + px],
            0, 0);
      }
    }
    __syncthreads();

    v8f acc[4];
#pragma unroll
    for (int i = 0; i < 4; ++i) acc[i] = vzero8();
    const _Float16* arow = As + ((size_t)wave * 16 + ln) * 288;
    for (int tap = 0; tap < 9; ++tap) {  // K = tap*32 + ch
      int dy = tap / 3 - 1, dx = tap % 3 - 1;
      v16h af = load_afrag(arow, tap * 32, hi);
      v16h bf[4];
#pragma unroll
      for (int nt = 0; nt < 4; ++nt) {
        int pix = nt * 16 + ln;
        int py = pix >> 3, px = pix & 7;
        bf[nt] = load_bfrag(halo + ((py + dy + 1) * 10 + (px + dx + 1)) * 32, 0, hi);
      }
#pragma unroll
      for (int nt = 0; nt < 4; ++nt) acc[nt] = wmma16(af, bf[nt], acc[nt]);
    }

    int p = wave / 4;  // 0:q 1:k 2:v
    const float* bias = (p == 0) ? bq : (p == 1) ? bk : bv;
#pragma unroll
    for (int nt = 0; nt < 4; ++nt) {
      int pix = nt * 16 + ln;
#pragma unroll
      for (int r = 0; r < 8; ++r) {
        int m = wave * 16 + r + 8 * hi;
        int co = m & 63;
        float val = acc[nt][r] + bias[co];
        int head = co >> 3, hcI = co & 7;
        int d = hcI * 64 + pix;
        size_t tn = (size_t)t * N_HEAD + head;
        if (p == 0)      qh[(tn * NWIN + rem) * DDIM + d] = (_Float16)val;
        else if (p == 1) kh[(tn * NWIN + rem) * DDIM + d] = (_Float16)val;
        else             vth[(tn * DDIM + d) * NWIN + rem] = (_Float16)val;
      }
    }
  }
}

// -------------------------------------------------------- k2: flash attention
// grid: (64 (t,head), 16 qblocks), 512 threads = 16 waves.
// wave = (wm 0..3) x (wd 0..3): wm -> 16 q rows, wd -> 128 d columns.
// LDS: Qs[64][512] Ks[64][512] Vts[512][64] Ps[64][64] f16 + 128 f32 (205KB)
__global__ void k2_attn(const _Float16* __restrict__ qh,
                        const _Float16* __restrict__ kh,
                        const _Float16* __restrict__ vth,
                        _Float16* __restrict__ yh) {
  extern __shared__ char smem_raw[];
  _Float16* Qs  = (_Float16*)smem_raw;     // [64][512]
  _Float16* Ks  = Qs + 64 * 512;           // [64][512]
  _Float16* Vts = Ks + 64 * 512;           // [512][64]
  _Float16* Ps  = Vts + 512 * 64;          // [64][64]
  float* Lsc  = (float*)(Ps + 64 * 64);    // [64] row rescale
  float* Lsum = Lsc + 64;                  // [64] row sum
  const int tid = threadIdx.x;
  const int lane = tid & 31, ln = lane & 15, hi = lane >> 4;
  const int wave = tid >> 5;
  const int wm = wave & 3, wd = wave >> 2;
  const int tn = blockIdx.x, t = tn >> 3, head = tn & 7;
  const int qb = blockIdx.y;
  const float sscale = 0.044194173824159216f;  // 1/sqrt(512)

  {  // Q block -> LDS (async)
    const _Float16* qsrc = qh + ((size_t)tn * NWIN + qb * 64) * DDIM;
    for (int c = tid; c < 64 * 512 / 8; c += 512)
      stage16(qsrc + c * 8, Qs + c * 8);
  }

  v8f o[8];
#pragma unroll
  for (int i = 0; i < 8; ++i) o[i] = vzero8();
  float mrow[8], lrow[8];
#pragma unroll
  for (int r = 0; r < 8; ++r) { mrow[r] = -1e30f; lrow[r] = 0.f; }

  for (int kb = 0; kb < 16; ++kb) {
    __syncthreads();
    {  // K block rows -> LDS (async)
      const _Float16* ksrc = kh + ((size_t)tn * NWIN + kb * 64) * DDIM;
      for (int c = tid; c < 64 * 512 / 8; c += 512)
        stage16(ksrc + c * 8, Ks + c * 8);
    }
    for (int c = tid; c < 512 * 8; c += 512) {  // V^T block (64 tokens per d row)
      int d = c >> 3, cc = c & 7;
      stage16(vth + ((size_t)tn * DDIM + d) * NWIN + kb * 64 + cc * 8,
              Vts + d * 64 + cc * 8);
    }
    stage_wait();
    __syncthreads();

    if (wd == 0) {  // wave-uniform branch: EXEC stays all-ones for WMMA
      v8f s[4];
#pragma unroll
      for (int i = 0; i < 4; ++i) s[i] = vzero8();
      const _Float16* qrow = Qs + (wm * 16 + ln) * DDIM;
      for (int kk = 0; kk < 16; ++kk) {
        v16h af = load_afrag(qrow, kk * 32, hi);
        v16h bf[4];
#pragma unroll
        for (int nt = 0; nt < 4; ++nt)  // B[k=d][n=token]
          bf[nt] = load_bfrag(Ks + (nt * 16 + ln) * DDIM, kk * 32, hi);
#pragma unroll
        for (int nt = 0; nt < 4; ++nt) s[nt] = wmma16(af, bf[nt], s[nt]);
      }
#pragma unroll
      for (int r = 0; r < 8; ++r) {  // online softmax per q row (row = r + 8*hi)
        float v0 = s[0][r] * sscale, v1 = s[1][r] * sscale;
        float v2 = s[2][r] * sscale, v3 = s[3][r] * sscale;
        float mx = fmaxf(fmaxf(v0, v1), fmaxf(v2, v3));
        mx = fmaxf(mx, __shfl_xor(mx, 1, 16));
        mx = fmaxf(mx, __shfl_xor(mx, 2, 16));
        mx = fmaxf(mx, __shfl_xor(mx, 4, 16));
        mx = fmaxf(mx, __shfl_xor(mx, 8, 16));
        float mnew = fmaxf(mrow[r], mx);
        float alpha = __expf(mrow[r] - mnew);
        float p0 = __expf(v0 - mnew), p1 = __expf(v1 - mnew);
        float p2 = __expf(v2 - mnew), p3 = __expf(v3 - mnew);
        float ps = p0 + p1 + p2 + p3;
        ps += __shfl_xor(ps, 1, 16);
        ps += __shfl_xor(ps, 2, 16);
        ps += __shfl_xor(ps, 4, 16);
        ps += __shfl_xor(ps, 8, 16);
        lrow[r] = lrow[r] * alpha + ps;
        mrow[r] = mnew;
        int q = wm * 16 + r + 8 * hi;
        Ps[q * 64 + 0 + ln]  = (_Float16)p0;
        Ps[q * 64 + 16 + ln] = (_Float16)p1;
        Ps[q * 64 + 32 + ln] = (_Float16)p2;
        Ps[q * 64 + 48 + ln] = (_Float16)p3;
        if (ln == 0) Lsc[q] = alpha;
      }
    }
    __syncthreads();

    // rescale accumulators and accumulate O += P * V
#pragma unroll
    for (int r = 0; r < 8; ++r) {
      float al = Lsc[wm * 16 + r + 8 * hi];
#pragma unroll
      for (int nt = 0; nt < 8; ++nt) o[nt][r] = o[nt][r] * al;
    }
    const _Float16* prow = Ps + (wm * 16 + ln) * 64;
#pragma unroll
    for (int kk = 0; kk < 2; ++kk) {
      v16h af = load_afrag(prow, kk * 32, hi);
      v16h bf[8];
#pragma unroll
      for (int nt = 0; nt < 8; ++nt)  // B[k=tok][n=d]
        bf[nt] = load_bfrag(Vts + (wd * 128 + nt * 16 + ln) * 64, kk * 32, hi);
#pragma unroll
      for (int nt = 0; nt < 8; ++nt) o[nt] = wmma16(af, bf[nt], o[nt]);
    }
  }

  if (wd == 0) {
#pragma unroll
    for (int r = 0; r < 8; ++r)
      if (ln == 0) Lsum[wm * 16 + r + 8 * hi] = lrow[r];
  }
  __syncthreads();
#pragma unroll
  for (int r = 0; r < 8; ++r) {
    int q = wm * 16 + r + 8 * hi;
    float inv = 1.0f / Lsum[q];
    int token = qb * 64 + q;
    int wy = token >> 5, wx = token & 31;
#pragma unroll
    for (int nt = 0; nt < 8; ++nt) {
      int d = wd * 128 + nt * 16 + ln;
      int c = head * 8 + (d >> 6);
      int pix = d & 63, py = pix >> 3, px = pix & 7;
      // channels-last y image for fast halo loads in k3
      size_t addr = (((size_t)t * IMG + wy * 8 + py) * IMG + wx * 8 + px) * C_OUT + c;
      yh[addr] = (_Float16)(o[nt][r] * inv);
    }
  }
}

// ------------------------------------------------- k3: output conv + residual
// grid: (16,16,8) tiles of 16x16 px; 256 threads = 8 waves.
// wave = (mt 0..3) x (nh 0..1): mt -> 16 out channels, nh -> 8 pixel n-tiles.
// LDS: Ws[64][576] (73.7KB) + halo[18][18][64] f16 (41.5KB)
__global__ void k3_outconv(const _Float16* __restrict__ yh,
                           const _Float16* __restrict__ woh,
                           const float* __restrict__ bo,
                           float* __restrict__ out) {
  extern __shared__ char smem_raw[];
  _Float16* Ws = (_Float16*)smem_raw;     // [64][576], k = tap*64+ch
  _Float16* halo = Ws + 64 * 576;         // [18][18][64]
  const int tid = threadIdx.x;
  const int lane = tid & 31, ln = lane & 15, hi = lane >> 4;
  const int wave = tid >> 5;
  const int mt = wave & 3, nh = wave >> 2;
  const int tx0 = blockIdx.x * 16, ty0 = blockIdx.y * 16;
  const int t = blockIdx.z;

  for (int c = tid; c < 64 * 576 / 8; c += 256)
    stage16(woh + c * 8, Ws + c * 8);
  // Halo: in-bounds cells via async load, out-of-bounds cells via ds zero-store.
  // The two paths touch disjoint LDS chunks (no DScnt/ASYNCcnt cross hazard).
  for (int c = tid; c < 18 * 18 * 8; c += 256) {  // 8 x 16B chunks per cell
    int cell = c >> 3, cc = c & 7;
    int hy = cell / 18, hx = cell % 18;
    int gy = ty0 + hy - 1, gx = tx0 + hx - 1;
    _Float16* ldst = halo + cell * 64 + cc * 8;
    if (gy >= 0 && gy < IMG && gx >= 0 && gx < IMG) {
      stage16(yh + (((size_t)t * IMG + gy) * IMG + gx) * C_OUT + cc * 8, ldst);
    } else {
      uint4 z = {0u, 0u, 0u, 0u};
      *(uint4*)ldst = z;
    }
  }
  stage_wait();
  __syncthreads();

  v8f acc[8];
#pragma unroll
  for (int i = 0; i < 8; ++i) acc[i] = vzero8();
  const _Float16* arow = Ws + (mt * 16 + ln) * 576;
  for (int kk = 0; kk < 18; ++kk) {  // K = 576 = 9 taps * 64 ch
    int tap = kk >> 1, chb = (kk & 1) * 32;
    int dy = tap / 3 - 1, dx = tap % 3 - 1;
    v16h af = load_afrag(arow, kk * 32, hi);
    v16h bf[8];
#pragma unroll
    for (int nt = 0; nt < 8; ++nt) {
      int p = (nh * 8 + nt) * 16 + ln;
      int py = p >> 4, px = p & 15;
      bf[nt] = load_bfrag(halo + ((py + dy + 1) * 18 + (px + dx + 1)) * 64 + chb,
                          0, hi);
    }
#pragma unroll
    for (int nt = 0; nt < 8; ++nt) acc[nt] = wmma16(af, bf[nt], acc[nt]);
  }

#pragma unroll
  for (int nt = 0; nt < 8; ++nt) {
    int p = (nh * 8 + nt) * 16 + ln;
    int py = p >> 4, px = p & 15;
    int gy = ty0 + py, gx = tx0 + px;
#pragma unroll
    for (int r = 0; r < 8; ++r) {
      int co = mt * 16 + r + 8 * hi;
      float resid = (float)yh[(((size_t)t * IMG + gy) * IMG + gx) * C_OUT + co];
      out[(((size_t)t * C_OUT + co) * IMG + gy) * IMG + gx] =
          resid + acc[nt][r] + bo[co];
    }
  }
}

// --------------------------------------------------------------------- launch
extern "C" void kernel_launch(void* const* d_in, const int* in_sizes, int n_in,
                              void* d_out, int out_size, void* d_ws, size_t ws_size,
                              hipStream_t stream) {
  const float* x  = (const float*)d_in[0];
  const float* Wq = (const float*)d_in[1];
  const float* bq = (const float*)d_in[2];
  const float* Wk = (const float*)d_in[3];
  const float* bk = (const float*)d_in[4];
  const float* Wv = (const float*)d_in[5];
  const float* bv = (const float*)d_in[6];
  const float* Wo = (const float*)d_in[7];
  const float* bo = (const float*)d_in[8];
  float* out = (float*)d_out;

  char* ws = (char*)d_ws;
  size_t off = 0;
  _Float16* wqkv = (_Float16*)(ws + off); off += (size_t)192 * 288 * 2;
  _Float16* woh  = (_Float16*)(ws + off); off += (size_t)64 * 576 * 2;
  _Float16* qh   = (_Float16*)(ws + off); off += (size_t)64 * NWIN * DDIM * 2;
  _Float16* kh   = (_Float16*)(ws + off); off += (size_t)64 * NWIN * DDIM * 2;
  _Float16* vth  = (_Float16*)(ws + off); off += (size_t)64 * DDIM * NWIN * 2;
  _Float16* yh   = (_Float16*)(ws + off); off += (size_t)T_DIM * IMG * IMG * C_OUT * 2;
  // total ~256 MB of d_ws

  size_t lds1 = (size_t)(192 * 288 + 10 * 10 * 32) * sizeof(_Float16);       // ~117KB
  size_t lds2 = (size_t)(64 * 512 * 3 + 64 * 64) * sizeof(_Float16) + 128 * 4;  // ~205KB
  size_t lds3 = (size_t)(64 * 576 + 18 * 18 * 64) * sizeof(_Float16);        // ~115KB
  (void)hipFuncSetAttribute((const void*)k1_qkv,
      hipFuncAttributeMaxDynamicSharedMemorySize, (int)lds1);
  (void)hipFuncSetAttribute((const void*)k2_attn,
      hipFuncAttributeMaxDynamicSharedMemorySize, (int)lds2);
  (void)hipFuncSetAttribute((const void*)k3_outconv,
      hipFuncAttributeMaxDynamicSharedMemorySize, (int)lds3);

  k0_pack_weights<<<(192 * 288 + 64 * 576 + 255) / 256, 256, 0, stream>>>(
      Wq, Wk, Wv, Wo, wqkv, woh);
  k1_qkv<<<8192 / WPB, 384, lds1, stream>>>(x, bq, bk, bv, wqkv, qh, kh, vth);
  dim3 g2(64, 16);
  k2_attn<<<g2, 512, lds2, stream>>>(qh, kh, vth, yh);
  dim3 g3(16, 16, 8);
  k3_outconv<<<g3, 256, lds3, stream>>>(yh, woh, bo, out);
}
